// Model_15710990369146
// MI455X (gfx1250) — matrix operand order repaired and checked by reading, not compiled, not run on hardware
//
#include <hip/hip_runtime.h>
#include <math.h>

#define B_      8
#define S_ENC_  3072
#define PRED_   1024
#define ENC_IN_ 40
#define D_      128
#define NHD_    4            // heads
#define BUCKET_ 64
#define NHASH_  4
#define NL_     2
#define DFF_    512
#define COUT_   4
#define NCLS_   3
#define MARK_   4
#define T_      (S_ENC_ + PRED_)   // 4096
#define DH_     (D_ / NHD_)        // 32
#define NBK_    (T_ / BUCKET_)     // 64 buckets per hash
#define NBTOT_  (NHASH_ * NBK_)    // 256 total buckets / chunks
#define BH_     (B_ * NHD_)        // 32
#define SELF_VAL (-5e4f)

typedef __attribute__((ext_vector_type(16))) _Float16 v16h;
typedef __attribute__((ext_vector_type(8)))  _Float16 v8h;
typedef __attribute__((ext_vector_type(8)))  float    v8f;
typedef __attribute__((ext_vector_type(4)))  float    v4f;
typedef __attribute__((ext_vector_type(4)))  unsigned int u32x4;
typedef __attribute__((ext_vector_type(8)))  int      i32x8;
typedef __attribute__((ext_vector_type(4)))  int      i32x4;

// ---------------------------------------------------------------------------
// WMMA fragment builders (layouts per CDNA5 ISA 7.12.2).
// A (16x32 f16): lane L -> M = L&15; K-base = (L&16)?8:0; per lane the 16
// elements are K = base..base+7 and base+16..base+23 (two contiguous runs).
// B (32x16 f16): lane L -> N = L&15; same K order as A (base..base+7, base+16..base+23).
// C/D (f32): lane L -> N = L&15; VGPR j -> M = j + ((L&16)?8:0).
// ---------------------------------------------------------------------------
__device__ __forceinline__ v16h combine88(v8h lo, v8h hi) {
  v16h a;
#pragma unroll
  for (int e = 0; e < 8; ++e) { a[e] = lo[e]; a[e + 8] = hi[e]; }
  return a;
}
// A fragment from f16 LDS, row-major [M x ld]
__device__ __forceinline__ v16h frag_a_lds_h(const _Float16* base, int ld, int m0, int k0, int lane) {
  int m = m0 + (lane & 15);
  int kb = k0 + ((lane & 16) ? 8 : 0);
  const _Float16* r = base + (size_t)m * ld + kb;
  v8h lo = *(const v8h*)r;
  v8h hi = *(const v8h*)(r + 16);
  return combine88(lo, hi);
}
// A fragment from f32 LDS, row-major [M x ld] (convert to f16)
__device__ __forceinline__ v16h frag_a_lds_f32(const float* base, int ld, int m0, int k0, int lane) {
  int m = m0 + (lane & 15);
  int kb = k0 + ((lane & 16) ? 8 : 0);
  const float* r = base + (size_t)m * ld + kb;
  v4f f0 = *(const v4f*)r;
  v4f f1 = *(const v4f*)(r + 4);
  v4f f2 = *(const v4f*)(r + 16);
  v4f f3 = *(const v4f*)(r + 20);
  v16h a;
#pragma unroll
  for (int e = 0; e < 4; ++e) {
    a[e] = (_Float16)f0[e];
    a[4 + e] = (_Float16)f1[e];
    a[8 + e] = (_Float16)f2[e];
    a[12 + e] = (_Float16)f3[e];
  }
  return a;
}
// B fragment from K-contiguous (n-major) f16 LDS: logical B[k][n] = src[n*ld + k]
__device__ __forceinline__ v16h frag_b_lds_kc(const _Float16* base, int ld, int n0, int k0, int lane) {
  int n = n0 + (lane & 15);
  int kb = k0 + ((lane & 16) ? 8 : 0);
  const _Float16* r = base + (size_t)n * ld + kb;
  return combine88(*(const v8h*)r, *(const v8h*)(r + 16));
}

// ---------------------------------------------------------------------------
// Tensor Data Mover: load a 2D f32 tile (tile_d0 x tile_d1 elements, row
// stride stride0 elements) from global into LDS at byte offset lds_off.
// D# per CDNA5 ISA 8.3/8.4 (count=1, type=2 image, data_size=4B).
// ---------------------------------------------------------------------------
__device__ __forceinline__ void tdm_load_tile_f32(const float* gsrc, unsigned lds_off,
                                                  int tensor_d0, int tensor_d1,
                                                  int tile_d0, int tile_d1, int stride0) {
  unsigned long long ga = (unsigned long long)(const void*)gsrc;
  u32x4 g0;
  g0[0] = 1u;                                             // count=1, user descriptor
  g0[1] = lds_off;                                        // lds_addr (bytes)
  g0[2] = (unsigned)(ga & 0xffffffffull);                 // global_addr[31:0]
  g0[3] = (unsigned)((ga >> 32) & 0x01ffffffu) | (2u << 30); // addr[56:32] | type=2
  i32x8 g1;
  g1[0] = (int)(2u << 16);                                // data_size=2 (4 bytes)
  g1[1] = (int)(((unsigned)tensor_d0 & 0xffffu) << 16);   // tensor_dim0[15:0]
  g1[2] = (int)((((unsigned)tensor_d0 >> 16) & 0xffffu) | (((unsigned)tensor_d1 & 0xffffu) << 16));
  g1[3] = (int)((((unsigned)tensor_d1 >> 16) & 0xffffu) | (((unsigned)tile_d0 & 0xffffu) << 16));
  g1[4] = (int)((unsigned)tile_d1 & 0xffffu);             // tile_dim1 (tile_dim2=0)
  g1[5] = stride0;                                        // tensor_dim0_stride[31:0]
  g1[6] = 0;
  g1[7] = 0;
  i32x4 g2; g2[0] = g2[1] = g2[2] = g2[3] = 0;
  i32x4 g3; g3[0] = g3[1] = g3[2] = g3[3] = 0;
#if defined(__clang_major__) && (__clang_major__ >= 23)
  i32x8 g4; g4[0]=g4[1]=g4[2]=g4[3]=g4[4]=g4[5]=g4[6]=g4[7]=0;
  __builtin_amdgcn_tensor_load_to_lds(g0, g1, g2, g3, g4, 0);
#else
  __builtin_amdgcn_tensor_load_to_lds(g0, g1, g2, g3, 0);
#endif
}

// ---------------- Generic WMMA GEMM: C[MxN] = A[MxK] @ B[KxN] (+bias)(+gelu) -
// Block = 256 threads (8 waves), tile 128(M) x 64(N), K-step 32.
// A panel staged to LDS by the Tensor Data Mover (f32), B panel staged
// transposed to K-contiguous f16 by vector loads.
#define GEMM_BIAS 1
#define GEMM_GELU 2
__global__ __launch_bounds__(256) void k_gemm(const float* __restrict__ A,
                                              const float* __restrict__ B,
                                              const float* __restrict__ bias,
                                              float* __restrict__ C,
                                              int M, int N, int K, int flags) {
  __shared__ alignas(32) float    As[128][32];   // 16 KB, row-major M x Kstep
  __shared__ alignas(32) _Float16 Bs[64][32];    // 4 KB, n-major (K-contiguous)
  const int tid = threadIdx.x;
  const int lane = tid & 31;
  const int wave = tid >> 5;
  const int m0g = blockIdx.y * 128;
  const int n0g = blockIdx.x * 64;
  const unsigned as_off = (unsigned)(unsigned long long)(uintptr_t)&As[0][0];

  v8f acc[4] = {};
  for (int k0 = 0; k0 < K; k0 += 32) {
    // --- stage A panel via TDM (wave 0 issues; DMA is a wave-level op) ---
    if (tid < 32) {
      tdm_load_tile_f32(A + (size_t)m0g * K + k0, as_off,
                        /*tensor_d0=*/K, /*tensor_d1=*/M - m0g,
                        /*tile_d0=*/32, /*tile_d1=*/128, /*stride0=*/K);
      __builtin_amdgcn_s_wait_tensorcnt(0);
    }
    // --- stage B panel (32 x 64 f32 -> Bs[n][k] f16), coalesced v4f loads ---
#pragma unroll
    for (int pass = 0; pass < 2; ++pass) {
      int idx = pass * 1024 + tid * 4;          // 2048 elements total
      int krow = idx >> 6;                      // 0..31
      int ncol = idx & 63;                      // multiple of 4
      v4f val = *(const v4f*)(B + (size_t)(k0 + krow) * N + n0g + ncol);
#pragma unroll
      for (int u = 0; u < 4; ++u) Bs[ncol + u][krow] = (_Float16)val[u];
    }
    __syncthreads();
    // --- compute: each wave owns a 16-row strip, 4 N-tiles ---
    v16h a = frag_a_lds_f32(&As[0][0], 32, wave * 16, 0, lane);
#pragma unroll
    for (int nt = 0; nt < 4; ++nt) {
      v16h b = frag_b_lds_kc(&Bs[0][0], 32, nt * 16, 0, lane);
      acc[nt] = __builtin_amdgcn_wmma_f32_16x16x32_f16(false, a, false, b,
                                                       (short)0, acc[nt], false, false);
    }
    __syncthreads();
  }
#pragma unroll
  for (int nt = 0; nt < 4; ++nt) {
    int n = n0g + nt * 16 + (lane & 15);
    float bv = (flags & GEMM_BIAS) ? bias[n] : 0.0f;
#pragma unroll
    for (int j = 0; j < 8; ++j) {
      int m = m0g + wave * 16 + j + ((lane & 16) ? 8 : 0);
      float r = acc[nt][j] + bv;
      if (flags & GEMM_GELU) r = 0.5f * r * (1.0f + erff(r * 0.70710678118654752f));
      C[(size_t)m * N + n] = r;
    }
  }
}

// ---------------- Embedding: circular conv + timeF linear + sinusoid PE -----
__global__ void k_embed(const float* __restrict__ x_enc, const float* __restrict__ x_dec,
                        const float* __restrict__ xm_enc, const float* __restrict__ xm_dec,
                        const float* __restrict__ conv_w, const float* __restrict__ timef_w,
                        float* __restrict__ h) {
  size_t idx = (size_t)blockIdx.x * blockDim.x + threadIdx.x;
  if (idx >= (size_t)B_ * T_ * D_) return;
  int d = idx % D_;
  size_t r = idx / D_;
  int t = (int)(r % T_);
  int b = (int)(r / T_);
  float s = 0.0f;
  for (int kk = 0; kk < 3; ++kk) {
    int ts = t - 1 + kk; ts = (ts + T_) % T_;
    const float* xr = (ts < S_ENC_) ? (x_enc + ((size_t)b * S_ENC_ + ts) * ENC_IN_)
                                    : (x_dec + ((size_t)b * PRED_ + (ts - S_ENC_)) * ENC_IN_);
    const float* wr = conv_w + ((size_t)kk * ENC_IN_) * D_ + d;
    for (int c = 0; c < ENC_IN_; ++c) s += xr[c] * wr[(size_t)c * D_];
  }
  const float* xmr = (t < S_ENC_) ? (xm_enc + ((size_t)b * S_ENC_ + t) * MARK_)
                                  : (xm_dec + ((size_t)b * PRED_ + (t - S_ENC_)) * MARK_);
  for (int mk = 0; mk < MARK_; ++mk) s += xmr[mk] * timef_w[mk * D_ + d];
  int i2 = d >> 1;
  float freq = expf(-(float)(2 * i2) * (logf(10000.0f) / (float)D_));
  float ang = (float)t * freq;
  s += (d & 1) ? cosf(ang) : sinf(ang);
  h[idx] = s;
}

// ---------------- LSH bucketing ---------------------------------------------
__global__ void k_buckets(const float* __restrict__ xqk, const float* __restrict__ rot_l,
                          int* __restrict__ bkt) {
  size_t idx = (size_t)blockIdx.x * blockDim.x + threadIdx.x;
  if (idx >= (size_t)BH_ * NHASH_ * T_) return;
  int t = (int)(idx % T_);
  int tmp = (int)(idx / T_);
  int hsh = tmp % NHASH_;
  int bh = tmp / NHASH_;
  int b = bh >> 2, hh = bh & 3;
  const float* q = xqk + ((size_t)(b * T_ + t)) * D_ + hh * DH_;
  float acc[32];
#pragma unroll
  for (int r0 = 0; r0 < 32; ++r0) acc[r0] = 0.0f;
  for (int d = 0; d < DH_; ++d) {
    float qd = q[d];
    const float* rr = rot_l + ((size_t)d * NHASH_ + hsh) * 32;
#pragma unroll
    for (int r0 = 0; r0 < 32; ++r0) acc[r0] += qd * rr[r0];
  }
  float best = -3.4e38f; int bi = 0;
  for (int r0 = 0; r0 < 64; ++r0) {
    float v = (r0 < 32) ? acc[r0] : -acc[r0 - 32];
    if (v > best) { best = v; bi = r0; }
  }
  bkt[(size_t)bh * (NHASH_ * T_) + (size_t)hsh * T_ + t] = bi + hsh * NBK_;
}

// ---------------- Stable counting sort of (bucket, t) -----------------------
__global__ void k_zero(int* p, int n) {
  int i = blockIdx.x * blockDim.x + threadIdx.x;
  if (i < n) p[i] = 0;
}
__global__ void k_count(const int* __restrict__ bkt, int* __restrict__ cnt) {
  size_t idx = (size_t)blockIdx.x * blockDim.x + threadIdx.x;
  if (idx >= (size_t)BH_ * NHASH_ * T_) return;
  int bh = (int)(idx / (NHASH_ * T_));
  atomicAdd(&cnt[bh * NBTOT_ + bkt[idx]], 1);
}
__global__ void k_scan(const int* __restrict__ cnt, int* __restrict__ offs) {
  int bh = blockIdx.x * blockDim.x + threadIdx.x;
  if (bh >= BH_) return;
  int s = 0;
  for (int u = 0; u < NBTOT_; ++u) { offs[bh * NBTOT_ + u] = s; s += cnt[bh * NBTOT_ + u]; }
}
__global__ void k_fill(const int* __restrict__ bkt, const int* __restrict__ offs,
                       int* __restrict__ st) {
  int idx = blockIdx.x * blockDim.x + threadIdx.x;  // bh*256 + bucket
  if (idx >= BH_ * NBTOT_) return;
  int bucket = idx & (NBTOT_ - 1);
  int bh = idx >> 8;
  int hsh = bucket >> 6;                 // bucket / NBK_
  int pos = offs[idx];
  const int* bb = bkt + (size_t)bh * (NHASH_ * T_) + (size_t)hsh * T_;
  int* so = st + (size_t)bh * (NHASH_ * T_);
  for (int t = 0; t < T_; ++t)
    if (bb[t] == bucket) so[pos++] = hsh * T_ + t;   // full ticker, stable in t
}

// ---------------- Chunked LSH attention (WMMA) ------------------------------
__global__ __launch_bounds__(128) void k_attn_chunk(const float* __restrict__ xqk,
                                                    const float* __restrict__ xv,
                                                    const int* __restrict__ st,
                                                    float* __restrict__ o_all,
                                                    float* __restrict__ lg) {
  const int chunk = blockIdx.x;       // 0..255
  const int bh = blockIdx.y;          // 0..31
  const int b = bh >> 2, hh = bh & 3;
  __shared__ alignas(32) _Float16 qs[64][DH_];      // 4 KB   (row-major q)
  __shared__ alignas(32) _Float16 ks[128][DH_];     // 8 KB   (key-major -> K-contig B)
  __shared__ alignas(32) _Float16 vsT[DH_][128];    // 8 KB   (d-major -> K-contig B)
  __shared__ alignas(32) float dots[64][128];       // 32 KB
  __shared__ int rowt[64];
  __shared__ int rowh[64];
  __shared__ int keyt[128];
  __shared__ float lses[64];
  const int tid = threadIdx.x;
  const int lane = tid & 31, wave = tid >> 5;
  const int* stb = st + (size_t)bh * (NHASH_ * T_);

  // gather q rows for this chunk
  if (tid < 64) {
    int tick = stb[chunk * 64 + tid];
    int t = tick & (T_ - 1);
    rowt[tid] = t;
    rowh[tid] = tick >> 12;             // tick / T_
    const float* src = xqk + ((size_t)(b * T_ + t)) * D_ + hh * DH_;
    for (int d = 0; d < DH_; ++d) qs[tid][d] = (_Float16)src[d];
  }
  // gather key/value rows: this chunk + previous chunk (look-back)
  {
    int prev = (chunk + NBTOT_ - 1) & (NBTOT_ - 1);
    int s = (tid < 64) ? (chunk * 64 + tid) : (prev * 64 + (tid - 64));
    int tick = stb[s];
    int t = tick & (T_ - 1);
    keyt[tid] = t;
    const float* srcq = xqk + ((size_t)(b * T_ + t)) * D_ + hh * DH_;
    const float* srcv = xv + ((size_t)(b * T_ + t)) * D_ + hh * DH_;
    float buf[DH_];
    float nrm = 0.0f;
    for (int d = 0; d < DH_; ++d) { buf[d] = srcq[d]; nrm += buf[d] * buf[d]; }
    nrm = fmaxf(sqrtf(nrm), 1e-12f);
    float inv = 1.0f / nrm;
    for (int d = 0; d < DH_; ++d) ks[tid][d] = (_Float16)(buf[d] * inv);
    for (int d = 0; d < DH_; ++d) vsT[d][tid] = (_Float16)srcv[d];
  }
  __syncthreads();

  // dots(64x128) = q(64x32) @ k^T(32x128); K = DH = 32 -> 1 WMMA per tile
  {
    int mi = wave;                      // 4 waves x 16 rows
    v16h a = frag_a_lds_h(&qs[0][0], DH_, mi * 16, 0, lane);
#pragma unroll
    for (int ni = 0; ni < 8; ++ni) {
      v16h bfr = frag_b_lds_kc(&ks[0][0], DH_, ni * 16, 0, lane);
      v8f acc = {};
      acc = __builtin_amdgcn_wmma_f32_16x16x32_f16(false, a, false, bfr,
                                                   (short)0, acc, false, false);
      int n = ni * 16 + (lane & 15);
#pragma unroll
      for (int j = 0; j < 8; ++j) {
        int m = mi * 16 + j + ((lane & 16) ? 8 : 0);
        dots[m][n] = acc[j];
      }
    }
  }
  __syncthreads();

  // scale + self-mask + softmax (logsumexp), p overwrites dots (f32)
  if (tid < 64) {
    const float scale = 0.17677669529663687f;   // DH^-0.5
    int tq = rowt[tid];
    float mx = -3.4e38f;
    for (int jj = 0; jj < 128; ++jj) {
      float dv = dots[tid][jj] * scale;
      if (keyt[jj] == tq) dv = SELF_VAL;
      dots[tid][jj] = dv;
      mx = fmaxf(mx, dv);
    }
    float sum = 0.0f;
    for (int jj = 0; jj < 128; ++jj) sum += expf(dots[tid][jj] - mx);
    float lse = mx + logf(sum);
    lses[tid] = lse;
    for (int jj = 0; jj < 128; ++jj) dots[tid][jj] = expf(dots[tid][jj] - lse);
  }
  __syncthreads();

  // o(64x32) = p(64x128) @ v(128x32); K loop of 4
  {
    int mi = wave;
#pragma unroll
    for (int ni = 0; ni < 2; ++ni) {
      v8f acc = {};
#pragma unroll
      for (int kc = 0; kc < 4; ++kc) {
        v16h a = frag_a_lds_f32(&dots[0][0], 128, mi * 16, kc * 32, lane);
        v16h bfr = frag_b_lds_kc(&vsT[0][0], 128, ni * 16, kc * 32, lane);
        acc = __builtin_amdgcn_wmma_f32_16x16x32_f16(false, a, false, bfr,
                                                     (short)0, acc, false, false);
      }
      int d = ni * 16 + (lane & 15);
#pragma unroll
      for (int j = 0; j < 8; ++j) {
        int i = mi * 16 + j + ((lane & 16) ? 8 : 0);
        int t = rowt[i], hsh = rowh[i];
        o_all[(((size_t)bh * NHASH_ + hsh) * T_ + t) * DH_ + d] = acc[j];
      }
    }
  }
  if (tid < 64)
    lg[((size_t)bh * NHASH_ + rowh[tid]) * T_ + rowt[tid]] = lses[tid];
}

// ---------------- Across-hash combine + head re-interleave ------------------
__global__ void k_combine(const float* __restrict__ o_all, const float* __restrict__ lg,
                          float* __restrict__ attn) {
  size_t idx = (size_t)blockIdx.x * blockDim.x + threadIdx.x;
  if (idx >= (size_t)BH_ * T_ * DH_) return;
  int d = (int)(idx % DH_);
  size_t r = idx / DH_;
  int t = (int)(r % T_);
  int bh = (int)(r / T_);
  float l[NHASH_];
  float mx = -3.4e38f;
  for (int hsh = 0; hsh < NHASH_; ++hsh) {
    l[hsh] = lg[((size_t)bh * NHASH_ + hsh) * T_ + t];
    mx = fmaxf(mx, l[hsh]);
  }
  float s = 0.0f;
  for (int hsh = 0; hsh < NHASH_; ++hsh) s += expf(l[hsh] - mx);
  float o = 0.0f;
  for (int hsh = 0; hsh < NHASH_; ++hsh)
    o += (expf(l[hsh] - mx) / s) * o_all[(((size_t)bh * NHASH_ + hsh) * T_ + t) * DH_ + d];
  int b = bh >> 2, hh = bh & 3;
  attn[((size_t)(b * T_ + t)) * D_ + hh * DH_ + d] = o;
}

// ---------------- Residual + LayerNorm (a may be null) ----------------------
__global__ void k_ln(const float* __restrict__ x, const float* __restrict__ a,
                     const float* __restrict__ g, const float* __restrict__ bta,
                     float* __restrict__ out) {
  int row = blockIdx.x * blockDim.x + threadIdx.x;
  if (row >= B_ * T_) return;
  const float* xr = x + (size_t)row * D_;
  const float* ar = a ? (a + (size_t)row * D_) : nullptr;
  float m = 0.0f;
  for (int d = 0; d < D_; ++d) m += xr[d] + (ar ? ar[d] : 0.0f);
  m *= (1.0f / D_);
  float var = 0.0f;
  for (int d = 0; d < D_; ++d) {
    float v = xr[d] + (ar ? ar[d] : 0.0f) - m;
    var += v * v;
  }
  var *= (1.0f / D_);
  float rstd = rsqrtf(var + 1e-5f);
  for (int d = 0; d < D_; ++d) {
    float v = xr[d] + (ar ? ar[d] : 0.0f);
    out[(size_t)row * D_ + d] = (v - m) * rstd * g[d] + bta[d];
  }
}

// ---------------- Final projection + classifier -----------------------------
__global__ void k_proj(const float* __restrict__ hfin, const float* __restrict__ pw,
                       const float* __restrict__ pb, float* __restrict__ out_ct) {
  int idx = blockIdx.x * blockDim.x + threadIdx.x;   // B*PRED*COUT
  if (idx >= B_ * PRED_ * COUT_) return;
  int c = idx & 3;
  int tp = (idx >> 2) % PRED_;
  int b = idx / (PRED_ * COUT_);
  const float* hr = hfin + ((size_t)(b * T_ + S_ENC_ + tp)) * D_;
  float s = pb[c];
  for (int d = 0; d < D_; ++d) s += hr[d] * pw[d * COUT_ + c];
  out_ct[idx] = s;
}

__global__ __launch_bounds__(128) void k_fc(const float* __restrict__ out_ct,
                                            const float* __restrict__ w,
                                            const float* __restrict__ bias,
                                            float* __restrict__ out) {
  int b = blockIdx.x;
  __shared__ float red[NCLS_][128];
  float acc[NCLS_] = {0.0f, 0.0f, 0.0f};
  const int n = PRED_ * COUT_;
  for (int i = threadIdx.x; i < n; i += 128) {
    float v = out_ct[(size_t)b * n + i];
    for (int c = 0; c < NCLS_; ++c) acc[c] += v * w[(size_t)i * NCLS_ + c];
  }
  for (int c = 0; c < NCLS_; ++c) red[c][threadIdx.x] = acc[c];
  __syncthreads();
  if (threadIdx.x == 0) {
    float lgt[NCLS_];
    for (int c = 0; c < NCLS_; ++c) {
      float s = bias[c];
      for (int k = 0; k < 128; ++k) s += red[c][k];
      lgt[c] = s;
    }
    float mx = fmaxf(lgt[0], fmaxf(lgt[1], lgt[2]));
    float sm = 0.0f;
    for (int c = 0; c < NCLS_; ++c) sm += expf(lgt[c] - mx);
    for (int c = 0; c < NCLS_; ++c) out[b * NCLS_ + c] = expf(lgt[c] - mx) / sm;
  }
}

// ---------------- Host launcher ---------------------------------------------
extern "C" void kernel_launch(void* const* d_in, const int* in_sizes, int n_in,
                              void* d_out, int out_size, void* d_ws, size_t ws_size,
                              hipStream_t stream) {
  const float* x_enc   = (const float*)d_in[0];
  const float* xm_enc  = (const float*)d_in[1];
  const float* x_dec   = (const float*)d_in[2];
  const float* xm_dec  = (const float*)d_in[3];
  const float* conv_w  = (const float*)d_in[4];
  const float* timef_w = (const float*)d_in[5];
  const float* Wqk     = (const float*)d_in[6];
  const float* Wv      = (const float*)d_in[7];
  const float* Wo      = (const float*)d_in[8];
  const float* bo      = (const float*)d_in[9];
  const float* rot     = (const float*)d_in[10];
  const float* ln1_g   = (const float*)d_in[11];
  const float* ln1_b   = (const float*)d_in[12];
  const float* ln2_g   = (const float*)d_in[13];
  const float* ln2_b   = (const float*)d_in[14];
  const float* w1      = (const float*)d_in[15];
  const float* b1      = (const float*)d_in[16];
  const float* w2      = (const float*)d_in[17];
  const float* b2      = (const float*)d_in[18];
  const float* lnf_g   = (const float*)d_in[19];
  const float* lnf_b   = (const float*)d_in[20];
  const float* proj_w  = (const float*)d_in[21];
  const float* proj_b  = (const float*)d_in[22];
  const float* fc1_w   = (const float*)d_in[23];
  const float* fc1_b   = (const float*)d_in[24];
  float* out = (float*)d_out;

  char* ws = (char*)d_ws;
  size_t off = 0;
  auto alloc = [&](size_t bytes) -> void* {
    off = (off + 255) & ~(size_t)255;
    void* p = ws + off;
    off += bytes;
    return p;
  };
  const size_t act = (size_t)B_ * T_ * D_ * sizeof(float);          // 16 MB
  float* h      = (float*)alloc(act);
  float* qkbuf  = (float*)alloc(act);
  float* vbuf   = (float*)alloc(act);
  float* attnb  = (float*)alloc(act);
  float* tmpb   = (float*)alloc(act);
  float* big    = (float*)alloc((size_t)BH_ * NHASH_ * T_ * DH_ * sizeof(float)); // 64MB: o_all / ffn1
  float* lg     = (float*)alloc((size_t)BH_ * NHASH_ * T_ * sizeof(float));
  int*   bkt    = (int*)alloc((size_t)BH_ * NHASH_ * T_ * sizeof(int));
  int*   st     = (int*)alloc((size_t)BH_ * NHASH_ * T_ * sizeof(int));
  int*   cnt    = (int*)alloc((size_t)BH_ * NBTOT_ * sizeof(int));
  int*   offs   = (int*)alloc((size_t)BH_ * NBTOT_ * sizeof(int));
  float* out_ct = (float*)alloc((size_t)B_ * PRED_ * COUT_ * sizeof(float));

  const int M = B_ * T_;   // 32768

  // embedding
  k_embed<<<(B_ * T_ * D_ + 255) / 256, 256, 0, stream>>>(x_enc, x_dec, xm_enc, xm_dec,
                                                          conv_w, timef_w, h);
  for (int l = 0; l < NL_; ++l) {
    const float* Wqk_l = Wqk + (size_t)l * D_ * D_;
    const float* Wv_l  = Wv + (size_t)l * D_ * D_;
    const float* Wo_l  = Wo + (size_t)l * D_ * D_;
    const float* bo_l  = bo + (size_t)l * D_;
    const float* rot_l = rot + (size_t)l * DH_ * NHASH_ * (NBK_ / 2);
    const float* w1_l  = w1 + (size_t)l * D_ * DFF_;
    const float* b1_l  = b1 + (size_t)l * DFF_;
    const float* w2_l  = w2 + (size_t)l * DFF_ * D_;
    const float* b2_l  = b2 + (size_t)l * D_;

    // qk / v projections
    k_gemm<<<dim3(D_ / 64, M / 128), 256, 0, stream>>>(h, Wqk_l, nullptr, qkbuf, M, D_, D_, 0);
    k_gemm<<<dim3(D_ / 64, M / 128), 256, 0, stream>>>(h, Wv_l, nullptr, vbuf, M, D_, D_, 0);

    // LSH buckets + stable counting sort
    k_buckets<<<(BH_ * NHASH_ * T_ + 255) / 256, 256, 0, stream>>>(qkbuf, rot_l, bkt);
    k_zero<<<(BH_ * NBTOT_ + 255) / 256, 256, 0, stream>>>(cnt, BH_ * NBTOT_);
    k_count<<<(BH_ * NHASH_ * T_ + 255) / 256, 256, 0, stream>>>(bkt, cnt);
    k_scan<<<1, BH_, 0, stream>>>(cnt, offs);
    k_fill<<<(BH_ * NBTOT_ + 255) / 256, 256, 0, stream>>>(bkt, offs, st);

    // chunked attention + hash combine
    k_attn_chunk<<<dim3(NBTOT_, BH_), 128, 0, stream>>>(qkbuf, vbuf, st, big, lg);
    k_combine<<<(BH_ * T_ * DH_ + 255) / 256, 256, 0, stream>>>(big, lg, attnb);

    // output projection, residual+LN
    k_gemm<<<dim3(D_ / 64, M / 128), 256, 0, stream>>>(attnb, Wo_l, bo_l, tmpb, M, D_, D_, GEMM_BIAS);
    k_ln<<<(M + 255) / 256, 256, 0, stream>>>(h, tmpb, ln1_g + l * D_, ln1_b + l * D_, h);

    // FFN (GELU fused), residual+LN
    k_gemm<<<dim3(DFF_ / 64, M / 128), 256, 0, stream>>>(h, w1_l, b1_l, big, M, DFF_, D_,
                                                         GEMM_BIAS | GEMM_GELU);
    k_gemm<<<dim3(D_ / 64, M / 128), 256, 0, stream>>>(big, w2_l, b2_l, tmpb, M, D_, DFF_, GEMM_BIAS);
    k_ln<<<(M + 255) / 256, 256, 0, stream>>>(h, tmpb, ln2_g + l * D_, ln2_b + l * D_, h);
  }

  // final LN, projection, classifier softmax
  k_ln<<<(M + 255) / 256, 256, 0, stream>>>(h, nullptr, lnf_g, lnf_b, h);
  k_proj<<<(B_ * PRED_ * COUT_ + 255) / 256, 256, 0, stream>>>(h, proj_w, proj_b, out_ct);
  k_fc<<<B_, 128, 0, stream>>>(out_ct, fc1_w, fc1_b, out);
}
